// SnakeScanBranch_65197603553689
// MI455X (gfx1250) — matrix operand order repaired and checked by reading, not compiled, not run on hardware
//
#include <hip/hip_runtime.h>
#include <math.h>

// ---------------- problem constants ----------------
#define Bn 8
#define Cn 128
#define Hn 128
#define Wn 128
#define HWn (Hn * Wn)
#define D_MODEL 128
#define D_INNER 256
#define D_STATE 64
#define HEADDIM 32
#define NHEADS 8
#define D_CONV 4
#define GN_GROUPS 8
#define D_XBC 384      // D_INNER + 2*D_STATE
#define D_INPROJ 648   // 2*D_INNER + 2*D_STATE + NHEADS
#define NSEQ (Bn * Hn) // 1024 sequences
#define Ln Wn          // sequence length 128

typedef unsigned int u32;
typedef unsigned short u16;
typedef unsigned long long u64;
typedef __attribute__((ext_vector_type(16))) __bf16 v16bf;
typedef __attribute__((ext_vector_type(8)))  float  v8f;
typedef __attribute__((ext_vector_type(4)))  u32    v4u;
// TDM descriptor groups (this toolchain uses the 6-arg builtin signature)
typedef __attribute__((ext_vector_type(4)))  u32    u32x4;
typedef __attribute__((ext_vector_type(8)))  int    i32x8;
typedef __attribute__((ext_vector_type(4)))  int    i32x4;

// ---------------- helpers ----------------
__device__ __forceinline__ float silu_f(float v) { return v / (1.f + __expf(-v)); }

// A-fragment (16x32 bf16, MxK): lane<16 holds K {0..7,16..23}, lane>=16 holds {8..15,24..31}
__device__ __forceinline__ v16bf load_frag_a(const u16* base /* row + ktile*32 */, int lane) {
    const int k0 = (lane < 16) ? 0 : 8;
    union { v16bf v; v4u q[2]; } u;
    u.q[0] = *(const v4u*)(base + k0);
    u.q[1] = *(const v4u*)(base + k0 + 16);
    return u.v;
}
// B-fragment (32x16 bf16, KxN): lane<16 holds K {0..7,16..23}, lane>=16 holds {8..15,24..31}
__device__ __forceinline__ v16bf load_frag_b(const u16* base /* colrow + ktile*32 */, int lane) {
    const int k0 = (lane < 16) ? 0 : 8;
    union { v16bf v; v4u q[2]; } u;
    u.q[0] = *(const v4u*)(base + k0);
    u.q[1] = *(const v4u*)(base + k0 + 16);
    return u.v;
}

// ---------------- K0: depthwise 5x5 conv (SAME) ----------------
__global__ void k_dwconv(const float* __restrict__ x, const float* __restrict__ dw,
                         float* __restrict__ h1) {
    const int bc = blockIdx.x;           // b*C*H + c*H + y
    const int b = bc / (Cn * Hn);
    const int rem = bc % (Cn * Hn);
    const int c = rem / Hn;
    const int y = rem % Hn;
    const int xc = threadIdx.x;          // 0..127
    const float* w = dw + c * 25;
    const float* xin = x + ((size_t)(b * Cn + c)) * HWn;
    float acc = 0.f;
    #pragma unroll
    for (int dy = -2; dy <= 2; ++dy) {
        const int yy = y + dy;
        if (yy < 0 || yy >= Hn) continue;
        #pragma unroll
        for (int dx = -2; dx <= 2; ++dx) {
            const int xx = xc + dx;
            if (xx < 0 || xx >= Wn) continue;
            acc += w[(dy + 2) * 5 + (dx + 2)] * xin[yy * Wn + xx];
        }
    }
    h1[((size_t)(b * Cn + c)) * HWn + y * Wn + xc] = acc;
}

// ---------------- K1: group-norm stats (64 blocks: b*8+g) ----------------
__global__ void k_gnstats(const float* __restrict__ h1, float* __restrict__ mu,
                          float* __restrict__ istd) {
    const int b = blockIdx.x / GN_GROUPS;
    const int g = blockIdx.x % GN_GROUPS;
    const int N = (Cn / GN_GROUPS) * HWn; // 262144
    __shared__ float s0[256], s1[256];
    float sum = 0.f, ssq = 0.f;
    const float* base = h1 + ((size_t)(b * Cn + g * (Cn / GN_GROUPS))) * HWn;
    for (int i = threadIdx.x; i < N; i += 256) {
        float v = base[i];
        sum += v; ssq += v * v;
    }
    s0[threadIdx.x] = sum; s1[threadIdx.x] = ssq;
    __syncthreads();
    for (int s = 128; s > 0; s >>= 1) {
        if (threadIdx.x < s) { s0[threadIdx.x] += s0[threadIdx.x + s]; s1[threadIdx.x] += s1[threadIdx.x + s]; }
        __syncthreads();
    }
    if (threadIdx.x == 0) {
        float m = s0[0] / (float)N;
        float v = s1[0] / (float)N - m * m;
        mu[blockIdx.x] = m;
        istd[blockIdx.x] = rsqrtf(v + 1e-5f);
    }
}

// ---------------- K2: GN + exact GELU + 1x1 conv + tanh -> offset ----------------
__global__ void k_offset(const float* __restrict__ h1, const float* __restrict__ mu,
                         const float* __restrict__ istd, const float* __restrict__ gn_g,
                         const float* __restrict__ gn_b, const float* __restrict__ pw_w,
                         const float* __restrict__ pw_b, float* __restrict__ out_off) {
    const int b = blockIdx.x / Hn;
    const int y = blockIdx.x % Hn;
    const int xc = threadIdx.x;
    float acc = 0.f;
    for (int c = 0; c < Cn; ++c) {
        const int g = c >> 4;
        float v = h1[((size_t)(b * Cn + c)) * HWn + y * Wn + xc];
        float vn = (v - mu[b * GN_GROUPS + g]) * istd[b * GN_GROUPS + g] * gn_g[c] + gn_b[c];
        float ge = 0.5f * vn * (1.f + erff(vn * 0.70710678118654752f));
        acc += pw_w[c] * ge;
    }
    float off = acc + pw_b[0];
    out_off[(b * Hn + y) * Wn + xc] = tanhf(off) * 8.0f; // min(8, 12.8) == 8
}

// ---------------- K3: vertical lerp (grid-sample) + pack bf16 sequences ----------------
__global__ void k_pack(const float* __restrict__ x, const float* __restrict__ off,
                       u16* __restrict__ seqbf) {
    const int b = blockIdx.x / Hn;
    const int y = blockIdx.x % Hn;
    const int xc = threadIdx.x;
    const float base_y = -1.f + 2.f * (float)y / (float)(Hn - 1);
    const float off_n = off[(b * Hn + y) * Wn + xc] * (2.f / (float)(Hn - 1));
    float gy = fminf(fmaxf(base_y + off_n, -1.f), 1.f);
    float gyp = fminf(fmaxf((gy + 1.f) * 0.5f * (float)(Hn - 1), 0.f), (float)(Hn - 1));
    float y0f = floorf(gyp);
    float wy = gyp - y0f;
    int y0 = (int)y0f;
    int y1 = min(y0 + 1, Hn - 1);
    __bf16* dst = (__bf16*)(seqbf + (((size_t)(b * Hn + y)) * Wn + xc) * Cn);
    for (int c = 0; c < Cn; ++c) {
        const float* im = x + ((size_t)(b * Cn + c)) * HWn;
        float v0 = im[y0 * Wn + xc];
        float v1 = im[y1 * Wn + xc];
        dst[c] = (__bf16)(v0 * (1.f - wy) + v1 * wy);
    }
}

// ---------------- K4: convert weights to bf16 (native v_cvt) ----------------
__global__ void k_cvtw(const float* __restrict__ inw, const float* __restrict__ ow,
                       u16* __restrict__ wbf, u16* __restrict__ owbf) {
    const int n1 = D_INPROJ * D_MODEL;        // 82944
    const int n2 = D_MODEL * D_INNER;         // 32768
    int i = blockIdx.x * blockDim.x + threadIdx.x;
    if (i < n1) ((__bf16*)wbf)[i] = (__bf16)inw[i];
    else if (i < n1 + n2) ((__bf16*)owbf)[i - n1] = (__bf16)ow[i - n1];
}

// ---------------- K5: fused Mamba block, one workgroup per sequence ----------------
// LDS layout (dynamic, 299008 B total):
//   [0,      32768)  Xl   : 128x128 bf16 input tile (loaded by TDM)
//   [32768, 98304)   Zl   : 128x256 bf16 (z, later reused for normalized y)
//   [98304, 294912)  XBC  : 128x384 f32 (xBC -> conv/silu -> x overwritten by ys)
//   [294912,299008)  DT   : 128x8 f32 (softplus dt)
__global__ void k_mamba(const u16* __restrict__ seqbf, const u16* __restrict__ wbf,
                        const u16* __restrict__ owbf, const float* __restrict__ in_proj_w,
                        const float* __restrict__ conv_w, const float* __restrict__ conv_b,
                        const float* __restrict__ dt_bias, const float* __restrict__ A_log,
                        const float* __restrict__ D_skip, const float* __restrict__ norm_g,
                        float* __restrict__ out2d) {
    extern __shared__ char smem[];
    u16*    Xl  = (u16*)smem;
    __bf16* Zl  = (__bf16*)(smem + 32768);
    float*  XBC = (float*)(smem + 98304);
    float*  DT  = (float*)(smem + 294912);

    const int tid  = threadIdx.x;
    const int lane = tid & 31;
    const int wv   = tid >> 5;        // 8 waves
    const int row  = blockIdx.x;      // 0..1023

    // hint the B-matrix weights into cache (gfx1250 global_prefetch_b8)
    __builtin_prefetch(wbf + (tid & 127) * 512, 0, 1);

    // ---- X tile (128 tokens x 128 ch bf16, 32KB contiguous) via Tensor Data Mover ----
    // D# per CDNA5 ISA: 2-D tensor, data_size=2B,
    // tensor_dim0 = tile_dim0 = 16384 elements, one row, stride 16384, count=1, type=2.
    if (wv == 0) {
        const u64 ga  = (u64)(size_t)(seqbf + (size_t)row * Ln * Cn);
        const u32 lds = (u32)(size_t)(void*)Xl;   // low 32 bits of flat addr = LDS byte offset
        u32x4 g0;
        g0.x = 1u;                                         // count=1, is_restore=0, gather off
        g0.y = lds;                                        // lds_addr [63:32]
        g0.z = (u32)ga;                                    // global_addr [95:64]
        g0.w = (u32)((ga >> 32) & 0x01FFFFFFu) | (2u << 30); // global_addr [120:96], type=2
        i32x8 g1;
        g1[0] = (int)(1u << 16);                           // workgroup_mask=0, data_size=1 (2B)
        g1[1] = (int)(0x4000u << 16);                      // tensor_dim0[15:0]=16384 at bits[63:48]
        g1[2] = (int)(1u << 16);                           // tensor_dim0[31:16]=0, tensor_dim1=1
        g1[3] = (int)(0x4000u << 16);                      // tensor_dim1 hi=0, tile_dim0=16384
        g1[4] = 1;                                         // tile_dim1=1, tile_dim2=0
        g1[5] = 16384;                                     // tensor_dim0_stride low 32
        g1[6] = 0;                                         // stride0 hi, stride1 low
        g1[7] = 0;
        i32x4 gz4 = {0, 0, 0, 0};                          // <=2-D: groups 2/3 unused
        i32x8 gz8 = {0, 0, 0, 0, 0, 0, 0, 0};
        __builtin_amdgcn_tensor_load_to_lds(g0, g1, gz4, gz4, gz8, 0);
    }
    __builtin_amdgcn_s_wait_tensorcnt(0);
    __syncthreads();

    // ---- dt = softplus(X @ Wdt^T + bias), VALU (8 cols only) ----
    #pragma unroll
    for (int i = 0; i < 4; ++i) {
        const int o = tid + 256 * i;          // 1024 outputs
        const int t = o >> 3, hh = o & 7;
        float acc = dt_bias[hh];
        const float* wr = in_proj_w + (size_t)(2 * D_INNER + 2 * D_STATE + hh) * D_MODEL;
        const __bf16* xr = (const __bf16*)Xl + t * Cn;
        for (int d = 0; d < D_MODEL; ++d) acc += (float)xr[d] * wr[d];
        DT[t * NHEADS + hh] = (acc > 20.f) ? acc : log1pf(__expf(acc));
    }

    // ---- phase-1 GEMM: zxbcdt[:, 0:640] via WMMA bf16 (wave = M-tile) ----
    for (int nt = 0; nt < 40; ++nt) {
        v8f acc = {0.f, 0.f, 0.f, 0.f, 0.f, 0.f, 0.f, 0.f};
        const int m = wv * 16 + (lane & 15);
        const int e = nt * 16 + (lane & 15);
        #pragma unroll
        for (int kt = 0; kt < 4; ++kt) {
            v16bf a = load_frag_a(Xl + m * Cn + kt * 32, lane);
            v16bf b = load_frag_b(wbf + e * D_MODEL + kt * 32, lane);
            acc = __builtin_amdgcn_wmma_f32_16x16x32_bf16(false, a, false, b,
                                                          (short)0, acc, false, false);
        }
        const int n = nt * 16 + (lane & 15);
        const int mbase = wv * 16 + ((lane < 16) ? 0 : 8);
        if (n < D_INNER) {
            #pragma unroll
            for (int r = 0; r < 8; ++r) Zl[(mbase + r) * D_INNER + n] = (__bf16)acc[r];
        } else {
            const int c = n - D_INNER;
            #pragma unroll
            for (int r = 0; r < 8; ++r) XBC[(mbase + r) * D_XBC + c] = acc[r];
        }
    }
    __syncthreads();

    // ---- causal conv-4 + SiLU, per channel, sliding raw window ----
    for (int c = tid; c < D_XBC; c += 256) {
        const float w0 = conv_w[c * 4 + 0], w1 = conv_w[c * 4 + 1];
        const float w2 = conv_w[c * 4 + 2], w3 = conv_w[c * 4 + 3];
        const float bb = conv_b[c];
        float p0 = 0.f, p1 = 0.f, p2 = 0.f;
        for (int t = 0; t < Ln; ++t) {
            float cur = XBC[t * D_XBC + c];
            float o = w0 * p0 + w1 * p1 + w2 * p2 + w3 * cur + bb;
            p0 = p1; p1 = p2; p2 = cur;
            XBC[t * D_XBC + c] = silu_f(o);
        }
    }
    __syncthreads();

    // ---- SSM scan: wave = head, lane = HEADDIM index, 64-deep state in VGPRs ----
    {
        const int h = wv, d = lane;
        const float Ah = -__expf(A_log[h]);
        const float Dh = D_skip[h];
        float st[D_STATE];
        #pragma unroll
        for (int s = 0; s < D_STATE; ++s) st[s] = 0.f;
        for (int t = 0; t < Ln; ++t) {
            const float dtv = DT[t * NHEADS + h];
            const float dA = __expf(dtv * Ah);
            const float xv = XBC[t * D_XBC + h * HEADDIM + d];
            const float coef = dtv * xv;
            float y = 0.f;
            const float4* Bp = (const float4*)(XBC + t * D_XBC + D_INNER);
            const float4* Cp = (const float4*)(XBC + t * D_XBC + D_INNER + D_STATE);
            #pragma unroll
            for (int s4 = 0; s4 < 16; ++s4) {
                float4 Bv = Bp[s4], Cv = Cp[s4];
                st[4*s4+0] = st[4*s4+0] * dA + coef * Bv.x; y += st[4*s4+0] * Cv.x;
                st[4*s4+1] = st[4*s4+1] * dA + coef * Bv.y; y += st[4*s4+1] * Cv.y;
                st[4*s4+2] = st[4*s4+2] * dA + coef * Bv.z; y += st[4*s4+2] * Cv.z;
                st[4*s4+3] = st[4*s4+3] * dA + coef * Bv.w; y += st[4*s4+3] * Cv.w;
            }
            XBC[t * D_XBC + h * HEADDIM + d] = y + Dh * xv; // overwrite x with ys
        }
    }
    __syncthreads();

    // ---- gate * SiLU(z), RMSNorm, write bf16 y over Zl ----
    for (int j = 0; j < 16; ++j) {
        const int t = wv + 8 * j;
        float yv[8], ss = 0.f;
        #pragma unroll
        for (int i = 0; i < 8; ++i) {
            const int c = lane + 32 * i;
            float ys = XBC[t * D_XBC + c];
            float zv = (float)Zl[t * D_INNER + c];
            float y = ys * silu_f(zv);
            yv[i] = y; ss += y * y;
        }
        #pragma unroll
        for (int m = 1; m < 32; m <<= 1) ss += __shfl_xor(ss, m, 32);
        const float rms = rsqrtf(ss * (1.f / (float)D_INNER) + 1e-5f);
        #pragma unroll
        for (int i = 0; i < 8; ++i) {
            const int c = lane + 32 * i;
            Zl[t * D_INNER + c] = (__bf16)(yv[i] * rms * norm_g[c]);
        }
    }
    __syncthreads();

    // ---- phase-2 GEMM: out[128x128] = Y(128x256) @ out_w^T, store NCHW ----
    const int b = row >> 7, yrow = row & 127;
    for (int nt = 0; nt < 8; ++nt) {
        v8f acc = {0.f, 0.f, 0.f, 0.f, 0.f, 0.f, 0.f, 0.f};
        const int m = wv * 16 + (lane & 15);
        const int dch = nt * 16 + (lane & 15);
        #pragma unroll
        for (int kt = 0; kt < 8; ++kt) {
            v16bf a = load_frag_a((const u16*)Zl + m * D_INNER + kt * 32, lane);
            v16bf bb = load_frag_b(owbf + dch * D_INNER + kt * 32, lane);
            acc = __builtin_amdgcn_wmma_f32_16x16x32_bf16(false, a, false, bb,
                                                          (short)0, acc, false, false);
        }
        const int c = nt * 16 + (lane & 15);
        const int mbase = wv * 16 + ((lane < 16) ? 0 : 8);
        #pragma unroll
        for (int r = 0; r < 8; ++r) {
            const int xcol = mbase + r;
            out2d[(((size_t)(b * Cn + c)) * Hn + yrow) * Wn + xcol] = acc[r];
        }
    }
}

// ---------------- host ----------------
extern "C" void kernel_launch(void* const* d_in, const int* in_sizes, int n_in,
                              void* d_out, int out_size, void* d_ws, size_t ws_size,
                              hipStream_t stream) {
    const float* x        = (const float*)d_in[0];
    const float* dw_w     = (const float*)d_in[1];
    const float* gn_g     = (const float*)d_in[2];
    const float* gn_b     = (const float*)d_in[3];
    const float* pw_w     = (const float*)d_in[4];
    const float* pw_b     = (const float*)d_in[5];
    const float* in_proj  = (const float*)d_in[6];
    const float* conv_w   = (const float*)d_in[7];
    const float* conv_b   = (const float*)d_in[8];
    const float* dt_bias  = (const float*)d_in[9];
    const float* A_log    = (const float*)d_in[10];
    const float* D_skip   = (const float*)d_in[11];
    const float* norm_g   = (const float*)d_in[12];
    const float* out_w    = (const float*)d_in[13];

    float* out2d   = (float*)d_out;
    float* out_off = out2d + (size_t)Bn * Cn * Hn * Wn;   // second tuple element

    char* ws = (char*)d_ws;
    float* h1    = (float*)ws;                              // 64 MiB
    u16*   seqbf = (u16*)(ws + (size_t)67108864);           // 32 MiB
    u16*   wbf   = (u16*)(ws + (size_t)100663296);          // 162 KiB
    u16*   owbf  = (u16*)(ws + (size_t)100829184);          // 64 KiB
    float* mu    = (float*)(ws + (size_t)100894720);        // 64 f
    float* istd  = mu + Bn * GN_GROUPS;

    k_dwconv<<<Bn * Cn * Hn, Wn, 0, stream>>>(x, dw_w, h1);
    k_gnstats<<<Bn * GN_GROUPS, 256, 0, stream>>>(h1, mu, istd);
    k_offset<<<Bn * Hn, Wn, 0, stream>>>(h1, mu, istd, gn_g, gn_b, pw_w, pw_b, out_off);
    k_pack<<<Bn * Hn, Wn, 0, stream>>>(x, out_off, seqbf);
    {
        const int tot = D_INPROJ * D_MODEL + D_MODEL * D_INNER;
        k_cvtw<<<(tot + 255) / 256, 256, 0, stream>>>(in_proj, out_w, wbf, owbf);
    }
    const int lds_bytes = 299008; // 292 KB, fits CDNA5's 320 KB/workgroup LDS
    (void)hipFuncSetAttribute((const void*)k_mamba,
                              hipFuncAttributeMaxDynamicSharedMemorySize, lds_bytes);
    k_mamba<<<NSEQ, 256, lds_bytes, stream>>>(seqbf, wbf, owbf, in_proj, conv_w, conv_b,
                                              dt_bias, A_log, D_skip, norm_g, out2d);
}
